// AttentionBlock_81338090652066
// MI455X (gfx1250) — matrix operand order repaired and checked by reading, not compiled, not run on hardware
//
#include <hip/hip_runtime.h>
#include <hip/hip_bf16.h>

// ---------------------------------------------------------------------------
// AttentionBlock on MI455X (gfx1250, wave32, WMMA f32_16x16x32_f16 + TDM)
// B=2, C=128, H=W=64 (L=4096), 4 heads (ch=32), GroupNormCustom + Rotary2D +
// flash-style attention (TDM-staged K/V tiles in LDS) + proj + residual.
// ---------------------------------------------------------------------------

typedef __attribute__((ext_vector_type(16))) _Float16 v16h;
typedef __attribute__((ext_vector_type(8)))  _Float16 v8h;
typedef __attribute__((ext_vector_type(8)))  float    v8f;
typedef __attribute__((ext_vector_type(4)))  float    v4f;
typedef __attribute__((ext_vector_type(4)))  unsigned int v4u;
typedef __attribute__((ext_vector_type(8)))  int      v8i;
typedef __attribute__((ext_vector_type(4)))  int      v4i;

constexpr int Bn = 2;
constexpr int Cn = 128;
constexpr int Ln = 4096;     // 64*64
constexpr int NH = 4;
constexpr int CH = 32;       // Cn / NH
constexpr int BH = Bn * NH;  // 8 batch-heads
constexpr float kQKScale = 0.42044820762685725f;  // 32^-0.25

__device__ __forceinline__ v8f zero8() {
  v8f z;
#pragma unroll
  for (int i = 0; i < 8; ++i) z[i] = 0.0f;
  return z;
}

__device__ __forceinline__ v8f wmma16(v16h a, v16h b, v8f c) {
  // D = A(16x32 f16) * B(32x16 f16) + C(16x16 f32)
  return __builtin_amdgcn_wmma_f32_16x16x32_f16(false, a, false, b, (short)0, c,
                                                false, false);
}

// A-operand (16x32 f16) from row-major storage, leading dim `ld` elements.
// ISA layout: lanes 0-15 -> rows 0-15 with K chunks {0..7, 16..23};
//             lanes 16-31 -> rows 0-15 with K chunks {8..15, 24..31}.
__device__ __forceinline__ v16h load_a16(const _Float16* base, int ld) {
  const int lane = threadIdx.x & 31;
  const _Float16* p = base + (lane & 15) * ld + ((lane & 16) ? 8 : 0);
  union { v16h v; v8h h[2]; } u;
  u.h[0] = *(const v8h*)(p);
  u.h[1] = *(const v8h*)(p + 16);
  return u.v;
}

// B-operand (32x16 f16): lane L holds column L%16, K chunks {0..7,16..23} / {8..15,24..31}.
__device__ __forceinline__ v16h load_b16(const _Float16* rowbase, int ld) {
  const int lane = threadIdx.x & 31;
  const _Float16* p = rowbase + (lane & 15) + ((lane & 16) ? 8 : 0) * ld;
  v16h b;
#pragma unroll
  for (int s = 0; s < 16; ++s) b[s] = p[((s < 8) ? s : (s + 8)) * ld];
  return b;
}

// B-operand from f32 rows (convert to f16 in VGPRs on the fly).
__device__ __forceinline__ v16h load_b_f32(const float* rowbase, int ld) {
  const int lane = threadIdx.x & 31;
  const float* p = rowbase + (lane & 15) + ((lane & 16) ? 8 : 0) * ld;
  v16h b;
#pragma unroll
  for (int s = 0; s < 16; ++s) b[s] = (_Float16)p[((s < 8) ? s : (s + 8)) * ld];
  return b;
}

// ---------------------------------------------------------------------------
// Tensor Data Mover: async DMA of a 2D tile (tile_x cols x tile_y rows of
// 2-byte elements, row stride `stride_e` elements) from global -> LDS.
// D# encoding per CDNA5 ISA (groups 0..3).
// Tracked by TENSORcnt; wait with __builtin_amdgcn_s_wait_tensorcnt.
// ---------------------------------------------------------------------------
__device__ __forceinline__ void tdm_load_2d_f16(const void* gptr, void* lptr,
                                                int tile_x, int tile_y,
                                                int stride_e) {
  unsigned long long ga = (unsigned long long)(uintptr_t)gptr;
  unsigned int la = (unsigned int)(uintptr_t)lptr;  // low 32 bits = LDS offset
  v4u g0;
  g0[0] = 1u;                                   // count=1, user D#, no gather
  g0[1] = la;                                   // lds_addr (bytes)
  g0[2] = (unsigned int)ga;                     // global_addr[31:0]
  g0[3] = (unsigned int)((ga >> 32) & 0x1FFFFFFu) | (2u << 30);  // [56:32]+type=2
  v8i g1;
  g1[0] = 1 << 16;            // workgroup_mask=0, data_size=1 (2 bytes)
  g1[1] = 0;                  // atomic_barrier off; tensor_dim0[15:0]=0 (dim0=1<<20)
  g1[2] = 0x0010;             // tensor_dim0[31:16]=0x10; tensor_dim1[15:0]=0
  g1[3] = (tile_x << 16) | 0x0010;  // tensor_dim1[31:16]=0x10; tile_dim0
  g1[4] = tile_y;             // tile_dim1; tile_dim2=0 (2D)
  g1[5] = stride_e;           // tensor_dim0_stride[31:0] (elements)
  g1[6] = 0;                  // stride[47:32]=0, tensor_dim1_stride lo=0
  g1[7] = 0;
  v4i g2;
  g2[0] = 0; g2[1] = 0; g2[2] = 0; g2[3] = 0;   // dims >2 unused
  v4i g3 = g2;
#if __clang_major__ >= 23
  v8i gz;
#pragma unroll
  for (int i = 0; i < 8; ++i) gz[i] = 0;
  __builtin_amdgcn_tensor_load_to_lds(g0, g1, g2, g3, gz, 0);
#else
  __builtin_amdgcn_tensor_load_to_lds(g0, g1, g2, g3, 0);
#endif
}

// ---------------------------------------------------------------------------
// Kernel 0: rotary sin/cos tables  (L x 64 each)
// freqs[l, i] = (i<32 ? h : w) * 10000^(-2*(i%32)/64),  l = h*64 + w
// ---------------------------------------------------------------------------
__global__ __launch_bounds__(256) void k_rope(float* sin_t, float* cos_t) {
  int idx = blockIdx.x * 256 + threadIdx.x;
  if (idx >= Ln * 64) return;
  int l = idx >> 6, i = idx & 63;
  int hh = l >> 6, ww = l & 63;
  int ii = (i < 32) ? i : (i - 32);
  float invf = __powf(10000.0f, -(2.0f * (float)ii) / 64.0f);
  float pos = (i < 32) ? (float)hh : (float)ww;
  float f = pos * invf;
  sin_t[idx] = __sinf(f);
  cos_t[idx] = __cosf(f);
}

// ---------------------------------------------------------------------------
// Kernel 1: GroupNormCustom stats: per (batch, j=c&3) over (H,W,groups),
// unbiased variance (ddof=1), eps=1e-6.
// stats layout: [b][0..3]=mean, [b][4..7]=rstd
// ---------------------------------------------------------------------------
__global__ __launch_bounds__(256) void k_gnstats(const float* __restrict__ x,
                                                 float* __restrict__ stats) {
  __shared__ float red[256];
  int b = blockIdx.x >> 2, j = blockIdx.x & 3;
  int tid = threadIdx.x;
  float s = 0.0f, s2 = 0.0f;
  for (int g = 0; g < 32; ++g) {
    const float* p = x + (size_t)(b * Cn + g * 4 + j) * Ln;
    for (int l = tid; l < Ln; l += 256) {
      float v = p[l];
      s += v;
      s2 += v * v;
    }
  }
  red[tid] = s;
  __syncthreads();
  for (int st = 128; st > 0; st >>= 1) {
    if (tid < st) red[tid] += red[tid + st];
    __syncthreads();
  }
  float tot = red[0];
  __syncthreads();
  red[tid] = s2;
  __syncthreads();
  for (int st = 128; st > 0; st >>= 1) {
    if (tid < st) red[tid] += red[tid + st];
    __syncthreads();
  }
  float tot2 = red[0];
  if (tid == 0) {
    const float n = 131072.0f;  // 64*64*32
    float mean = tot / n;
    float var = (tot2 - tot * tot / n) / (n - 1.0f);  // unbiased
    stats[b * 8 + j] = mean;
    stats[b * 8 + 4 + j] = __frsqrt_rn(var + 1e-6f);
  }
}

// ---------------------------------------------------------------------------
// Kernel 2: fold GroupNorm affine into QKV weights (per batch), proj weights
// to f16, and per-(b,c) residual coefficients: xi = x*acoef + bcoef.
// ---------------------------------------------------------------------------
__global__ __launch_bounds__(256) void k_prep(
    const float* __restrict__ w_qkv, const float* __restrict__ b_qkv,
    const float* __restrict__ gn_w, const float* __restrict__ gn_b,
    const float* __restrict__ w_proj, const float* __restrict__ stats,
    _Float16* __restrict__ w_eff, float* __restrict__ bias_eff,
    _Float16* __restrict__ w_proj_h, float* __restrict__ acoef,
    float* __restrict__ bcoef) {
  int t = blockIdx.x * 256 + threadIdx.x;
  if (t < Bn * Cn) {
    int b = t >> 7, c = t & 127;
    float mean = stats[b * 8 + (c & 3)];
    float rstd = stats[b * 8 + 4 + (c & 3)];
    float a = rstd * gn_w[c];
    acoef[t] = a;
    bcoef[t] = gn_b[c] - mean * a;
  }
  if (t < Cn * Cn) w_proj_h[t] = (_Float16)w_proj[t];
  if (t < Bn * 3 * Cn) {
    int b = t / (3 * Cn), o = t % (3 * Cn);
    float bias = b_qkv[o];
    for (int c = 0; c < Cn; ++c) {
      float mean = stats[b * 8 + (c & 3)];
      float rstd = stats[b * 8 + 4 + (c & 3)];
      float a = rstd * gn_w[c];
      float bb = gn_b[c] - mean * a;
      float w = w_qkv[o * Cn + c];
      w_eff[(size_t)(b * 3 * Cn + o) * Cn + c] = (_Float16)(w * a);
      bias += w * bb;
    }
    bias_eff[t] = bias;
  }
}

// ---------------------------------------------------------------------------
// Kernel 3: QKV GEMM (WMMA) + bias + Rotary2D + qk scale.
// Outputs: qs/vs as (bh, l, 32) f16, ks as (bh, 32, L) f16.
// ---------------------------------------------------------------------------
__global__ __launch_bounds__(256) void k_qkv(
    const float* __restrict__ x, const _Float16* __restrict__ w_eff,
    const float* __restrict__ bias_eff, const float* __restrict__ sin_t,
    const float* __restrict__ cos_t, _Float16* __restrict__ qs,
    _Float16* __restrict__ ks, _Float16* __restrict__ vs) {
  int wave = (blockIdx.x * 256 + threadIdx.x) >> 5;
  int lane = threadIdx.x & 31;
  int b = wave / (24 * 256);
  int rem = wave % (24 * 256);
  int o0 = (rem / 256) * 16;
  int l0 = (rem % 256) * 16;

  v8f acc = zero8();
#pragma unroll
  for (int kk = 0; kk < 4; ++kk) {
    int c0 = kk * 32;
    v16h A = load_a16(w_eff + (size_t)(b * 384 + o0) * Cn + c0, Cn);
    v16h Bf = load_b_f32(x + (size_t)(b * Cn + c0) * Ln + l0, Ln);
    acc = wmma16(A, Bf, acc);
  }

  // ---- epilogue: bias + rotary + scale -> f16 stores -------------------
  int n = lane & 15, hi = lane >> 4;
  int l = l0 + n;
  const float* bp = bias_eff + b * 384 + o0 + 8 * hi;
  bool isqk = (o0 < 256);
  v4f s4, c4;
  if (isqk) {
    int ib = ((o0 & 127) >> 1) + 4 * hi;  // rotary pair index base
    s4 = *(const v4f*)(sin_t + l * 64 + ib);
    c4 = *(const v4f*)(cos_t + l * 64 + ib);
  }
  _Float16 h[8];
#pragma unroll
  for (int j = 0; j < 4; ++j) {
    float e = acc[2 * j] + bp[2 * j];
    float o_ = acc[2 * j + 1] + bp[2 * j + 1];
    if (isqk) {
      float sn = s4[j], cs = c4[j];
      float re = e * cs - o_ * sn;   // even:  t*cos - t_odd*sin
      float ro = o_ * cs + e * sn;   // odd :  t*cos + t_even*sin
      e = re * kQKScale;
      o_ = ro * kQKScale;
    }
    h[2 * j] = (_Float16)e;
    h[2 * j + 1] = (_Float16)o_;
  }
  union { v8h v; _Float16 e[8]; } u;
#pragma unroll
  for (int r = 0; r < 8; ++r) u.e[r] = h[r];

  if (o0 < 128) {                       // Q -> (bh, l, ch)
    int ob = o0 + 8 * hi;
    int bh = b * NH + (ob >> 5), chh = ob & 31;
    *(v8h*)(qs + (size_t)(bh * Ln + l) * CH + chh) = u.v;
  } else if (o0 < 256) {                // K -> (bh, ch, L)
    int ob = o0 - 128 + 8 * hi;
    int bh = b * NH + (ob >> 5), chh = ob & 31;
#pragma unroll
    for (int r = 0; r < 8; ++r)
      ks[(size_t)(bh * CH + chh + r) * Ln + l] = h[r];
  } else {                              // V -> (bh, l, ch)
    int ob = o0 - 256 + 8 * hi;
    int bh = b * NH + (ob >> 5), chh = ob & 31;
    *(v8h*)(vs + (size_t)(bh * Ln + l) * CH + chh) = u.v;
  }
}

// ---------------------------------------------------------------------------
// Kernel 4: flash attention with TDM-staged K/V.
// Block = 8 waves, one batch-head, 8 consecutive 16-query tiles. Per 32-key
// iteration: wave 0 TDM-issues next K(32ch x 32key) and V(32key x 32ch) tiles
// into the other LDS buffer (async, TENSORcnt), all waves compute on the
// current buffer: 2 WMMA (S), online f32 softmax, P->LDS->A-layout, 2 WMMA
// (PV). Output a = (b, c, L) f16.
// ---------------------------------------------------------------------------
__global__ __launch_bounds__(256) void k_attn(const _Float16* __restrict__ qs,
                                              const _Float16* __restrict__ ks,
                                              const _Float16* __restrict__ vs,
                                              _Float16* __restrict__ as) {
  __shared__ alignas(16) _Float16 kT[2][32 * 32];  // (ch, key) tiles, 2KB each
  __shared__ alignas(16) _Float16 vT[2][32 * 32];  // (key, ch) tiles
  __shared__ alignas(16) _Float16 pt[8][16 * 32];  // per-wave P tiles

  int widx = threadIdx.x >> 5, lane = threadIdx.x & 31;
  int bh = blockIdx.x >> 5;
  int q0 = (blockIdx.x & 31) * 128 + widx * 16;
  int n = lane & 15, hi = lane >> 4;
  _Float16* myp = pt[widx];

  v16h Aq = load_a16(qs + (size_t)(bh * Ln + q0) * CH, CH);
  v8f O0 = zero8(), O1 = zero8();
  float m[8], lsum[8];
#pragma unroll
  for (int r = 0; r < 8; ++r) {
    m[r] = -3.0e38f;
    lsum[r] = 0.0f;
  }

  const _Float16* kg = ks + (size_t)bh * CH * Ln;   // rows=ch (stride Ln)
  const _Float16* vg = vs + (size_t)bh * Ln * CH;   // rows=key (stride CH)

  if (widx == 0) {
    tdm_load_2d_f16(kg, kT[0], 32, 32, Ln);
    tdm_load_2d_f16(vg, vT[0], 32, 32, CH);
    __builtin_amdgcn_s_wait_tensorcnt((short)0);
  }
  __syncthreads();

  for (int kt = 0; kt < 128; ++kt) {
    int cur = kt & 1, nxt = cur ^ 1;
    if (widx == 0 && kt + 1 < 128) {   // async prefetch of next tiles
      tdm_load_2d_f16(kg + (kt + 1) * 32, kT[nxt], 32, 32, Ln);
      tdm_load_2d_f16(vg + (size_t)(kt + 1) * 32 * CH, vT[nxt], 32, 32, CH);
    }

    v16h Bk0 = load_b16(kT[cur], 32);
    v16h Bk1 = load_b16(kT[cur] + 16, 32);
    v8f S0 = wmma16(Aq, Bk0, zero8());
    v8f S1 = wmma16(Aq, Bk1, zero8());

    // ---- online softmax, rows live in 16-lane groups ----
#pragma unroll
    for (int r = 0; r < 8; ++r) {
      float t = fmaxf(S0[r], S1[r]);
      t = fmaxf(t, __shfl_xor(t, 1, 32));
      t = fmaxf(t, __shfl_xor(t, 2, 32));
      t = fmaxf(t, __shfl_xor(t, 4, 32));
      t = fmaxf(t, __shfl_xor(t, 8, 32));
      float mn = fmaxf(m[r], t);
      float corr = __expf(m[r] - mn);
      m[r] = mn;
      float p0 = __expf(S0[r] - mn);
      float p1 = __expf(S1[r] - mn);
      float rs = p0 + p1;
      rs += __shfl_xor(rs, 1, 32);
      rs += __shfl_xor(rs, 2, 32);
      rs += __shfl_xor(rs, 4, 32);
      rs += __shfl_xor(rs, 8, 32);
      lsum[r] = lsum[r] * corr + rs;
      O0[r] *= corr;
      O1[r] *= corr;
      // stash P tile row-major (q, key): q = 8*hi + r; keys n and n+16
      int q = 8 * hi + r;
      myp[q * 32 + n] = (_Float16)p0;
      myp[q * 32 + 16 + n] = (_Float16)p1;
    }
    // Same-wave DS ops complete in order (ISA §7.3); fence scheduler and
    // drain DScnt before re-reading the tile in A-operand layout.
    __builtin_amdgcn_wave_barrier();
    asm volatile("s_wait_dscnt 0x0" ::: "memory");

    v16h Ap = load_a16(myp, 32);
    v16h Bv0 = load_b16(vT[cur], 32);
    v16h Bv1 = load_b16(vT[cur] + 16, 32);
    O0 = wmma16(Ap, Bv0, O0);
    O1 = wmma16(Ap, Bv1, O1);

    if (widx == 0 && kt + 1 < 128)
      __builtin_amdgcn_s_wait_tensorcnt((short)0);
    __syncthreads();  // publish next buffer / protect cur from overwrite
  }

  // ---- normalize rows and store as (b, c, L) f16 ----
  int b = bh >> 2, head = bh & 3;
  v8h r0, r1;
#pragma unroll
  for (int r = 0; r < 8; ++r) {
    float inv = 1.0f / lsum[r];
    r0[r] = (_Float16)(O0[r] * inv);
    r1[r] = (_Float16)(O1[r] * inv);
  }
  *(v8h*)(as + (size_t)(b * Cn + head * CH + n) * Ln + q0 + 8 * hi) = r0;
  *(v8h*)(as + (size_t)(b * Cn + head * CH + 16 + n) * Ln + q0 + 8 * hi) = r1;
}

// ---------------------------------------------------------------------------
// Kernel 5: output projection (WMMA) + bias + GroupNorm residual.
// out[b,o,l] = sum_c w_proj[o,c]*a[b,c,l] + b_proj[o] + (x*acoef + bcoef)
// ---------------------------------------------------------------------------
__global__ __launch_bounds__(256) void k_proj(
    const _Float16* __restrict__ as, const _Float16* __restrict__ w_proj_h,
    const float* __restrict__ b_proj, const float* __restrict__ x,
    const float* __restrict__ acoef, const float* __restrict__ bcoef,
    float* __restrict__ out) {
  int wave = (blockIdx.x * 256 + threadIdx.x) >> 5;
  int lane = threadIdx.x & 31;
  int b = wave / (8 * 256);
  int rem = wave % (8 * 256);
  int o0 = (rem / 256) * 16;
  int l0 = (rem % 256) * 16;

  v8f acc = zero8();
#pragma unroll
  for (int kk = 0; kk < 4; ++kk) {
    int c0 = kk * 32;
    v16h A = load_a16(w_proj_h + (size_t)o0 * Cn + c0, Cn);
    v16h Bf = load_b16(as + (size_t)(b * Cn + c0) * Ln + l0, Ln);
    acc = wmma16(A, Bf, acc);
  }

  int n = lane & 15, hi = lane >> 4;
  int l = l0 + n;
#pragma unroll
  for (int r = 0; r < 8; ++r) {
    int o = o0 + 8 * hi + r;
    size_t idx = (size_t)(b * Cn + o) * Ln + l;
    float xi = x[idx] * acoef[b * Cn + o] + bcoef[b * Cn + o];
    out[idx] = acc[r] + b_proj[o] + xi;
  }
}

// ---------------------------------------------------------------------------
extern "C" void kernel_launch(void* const* d_in, const int* in_sizes, int n_in,
                              void* d_out, int out_size, void* d_ws,
                              size_t ws_size, hipStream_t stream) {
  (void)in_sizes; (void)n_in; (void)out_size; (void)ws_size;
  const float* x      = (const float*)d_in[0];
  const float* gn_w   = (const float*)d_in[1];
  const float* gn_b   = (const float*)d_in[2];
  const float* w_qkv  = (const float*)d_in[3];
  const float* b_qkv  = (const float*)d_in[4];
  const float* w_proj = (const float*)d_in[5];
  const float* b_proj = (const float*)d_in[6];
  float* out = (float*)d_out;

  // ---- workspace carve-up (256B aligned), ~10.7 MB total ----
  char* ws = (char*)d_ws;
  size_t off = 0;
  auto carve = [&](size_t bytes) -> void* {
    void* p = (void*)(ws + off);
    off = (off + bytes + 255) & ~(size_t)255;
    return p;
  };
  float*    stats    = (float*)carve(16 * sizeof(float));
  float*    sin_t    = (float*)carve((size_t)Ln * 64 * sizeof(float));
  float*    cos_t    = (float*)carve((size_t)Ln * 64 * sizeof(float));
  float*    acoef    = (float*)carve(Bn * Cn * sizeof(float));
  float*    bcoef    = (float*)carve(Bn * Cn * sizeof(float));
  float*    bias_eff = (float*)carve(Bn * 3 * Cn * sizeof(float));
  _Float16* w_eff    = (_Float16*)carve((size_t)Bn * 3 * Cn * Cn * 2);
  _Float16* w_projh  = (_Float16*)carve((size_t)Cn * Cn * 2);
  _Float16* qs       = (_Float16*)carve((size_t)BH * Ln * CH * 2);
  _Float16* ks       = (_Float16*)carve((size_t)BH * CH * Ln * 2);
  _Float16* vs       = (_Float16*)carve((size_t)BH * Ln * CH * 2);
  _Float16* as       = (_Float16*)carve((size_t)Bn * Cn * Ln * 2);

  k_rope<<<(Ln * 64 + 255) / 256, 256, 0, stream>>>(sin_t, cos_t);
  k_gnstats<<<Bn * 4, 256, 0, stream>>>(x, stats);
  k_prep<<<(Cn * Cn + 255) / 256, 256, 0, stream>>>(
      w_qkv, b_qkv, gn_w, gn_b, w_proj, stats, w_eff, bias_eff, w_projh, acoef,
      bcoef);
  // 2 batches * 24 o-tiles * 256 l-tiles waves, 8 waves/block
  k_qkv<<<(Bn * 24 * 256) / 8, 256, 0, stream>>>(x, w_eff, bias_eff, sin_t,
                                                 cos_t, qs, ks, vs);
  // 8 bh * 32 query-blocks (128 queries each, 8 waves)
  k_attn<<<BH * 32, 256, 0, stream>>>(qs, ks, vs, as);
  // 2 batches * 8 o-tiles * 256 l-tiles waves, 8 waves/block
  k_proj<<<(Bn * 8 * 256) / 8, 256, 0, stream>>>(as, w_projh, b_proj, x, acoef,
                                                 bcoef, out);
}
